// EncoderBlock_80272938762587
// MI455X (gfx1250) — matrix operand order repaired and checked by reading, not compiled, not run on hardware
//
#include <hip/hip_runtime.h>

// ---------------------------------------------------------------------------
// MI455X (gfx1250) transformer encoder block.
// All GEMM-shaped work runs on v_wmma_f32_16x16x32_bf16 (wave32, 16x16x32).
// fp32 inputs are cast to bf16 for the matrix units; accumulation is fp32.
//
// Roofline: ~275 GFLOP total; intermediates (~200MB) mostly resident in the
// 192MB L2. Dense-bf16 matrix throughput bound -> maximize WMMA issue rate:
//  - 128x128 block tiles, 32x64 per wave (8 WMMA accums), K-step 32
//  - double-buffered LDS staging, one s_barrier per K-step
//  - bank-conflict-free LDS strides (17 dwords)
//  - global_prefetch_b8 for the K+2 tile
//  - softmax row-sums computed on the matrix pipe (P @ ones) instead of
//    ds_bpermute shuffle chains
// ---------------------------------------------------------------------------

typedef __bf16 bf16_t;
typedef bf16_t v16bf __attribute__((ext_vector_type(16)));
typedef float  v8f   __attribute__((ext_vector_type(8)));

#define D_MODEL 1024
#define D_FF    4096
#define NH      16
#define BATCH   4
#define SEQ     2048
#define DK      64
#define ROWS    (BATCH * SEQ)   // 8192

__device__ __forceinline__ unsigned short f2bf(float f) {
  unsigned int u = __float_as_uint(f);
  u += 0x7FFFu + ((u >> 16) & 1u);           // round to nearest even
  return (unsigned short)(u >> 16);
}

__device__ __forceinline__ unsigned int pack2bf(float lo, float hi) {
  return (unsigned int)f2bf(lo) | ((unsigned int)f2bf(hi) << 16);
}

__device__ __forceinline__ v8f wmma_bf16(v16bf a, v16bf b, v8f c) {
  // 8 args: (neg_a, A, neg_b, B, c_mod, C, reuse_a, reuse_b)
  return __builtin_amdgcn_wmma_f32_16x16x32_bf16(false, a, false, b, (short)0, c,
                                                 false, false);
}

// ---------------------------------------------------------------------------
// fp32 -> bf16 cast (weights), float4 in / uint2 out.
// ---------------------------------------------------------------------------
__global__ void cast_f32_bf16_kernel(const float4* __restrict__ src,
                                     uint2* __restrict__ dst, int n4) {
  int i = blockIdx.x * blockDim.x + threadIdx.x;
  int stride = gridDim.x * blockDim.x;
  for (; i < n4; i += stride) {
    float4 f = src[i];
    uint2 r;
    r.x = pack2bf(f.x, f.y);
    r.y = pack2bf(f.z, f.w);
    dst[i] = r;
  }
}

// ---------------------------------------------------------------------------
// Custom LayerNorm: alpha*(x-mean)/(std+eps)+beta, unbiased std (N-1).
// One block (8 waves) per row of 1024; float4 loads, uint2 bf16 stores.
// ---------------------------------------------------------------------------
__global__ __launch_bounds__(256)
void ln_bf16_kernel(const float* __restrict__ x,
                    const float* __restrict__ alpha,
                    const float* __restrict__ beta,
                    unsigned short* __restrict__ out) {
  __shared__ float s_sum[8];
  __shared__ float s_sq[8];
  const int row = blockIdx.x;
  const float4 v4 = ((const float4*)(x + (size_t)row * D_MODEL))[threadIdx.x];
  float sum = v4.x + v4.y + v4.z + v4.w;
  float sq  = v4.x * v4.x + v4.y * v4.y + v4.z * v4.z + v4.w * v4.w;
#pragma unroll
  for (int off = 16; off > 0; off >>= 1) {
    sum += __shfl_xor(sum, off, 32);
    sq  += __shfl_xor(sq,  off, 32);
  }
  const int wave = threadIdx.x >> 5;
  const int lane = threadIdx.x & 31;
  if (lane == 0) { s_sum[wave] = sum; s_sq[wave] = sq; }
  __syncthreads();
  sum = 0.f; sq = 0.f;
#pragma unroll
  for (int w = 0; w < 8; ++w) { sum += s_sum[w]; sq += s_sq[w]; }
  const float mean = sum * (1.0f / D_MODEL);
  const float var  = (sq - sum * mean) * (1.0f / (D_MODEL - 1));
  const float inv  = alpha[0] / (sqrtf(var) + 1e-5f);
  const float b0   = beta[0];
  uint2 o;
  o.x = pack2bf((v4.x - mean) * inv + b0, (v4.y - mean) * inv + b0);
  o.y = pack2bf((v4.z - mean) * inv + b0, (v4.w - mean) * inv + b0);
  ((uint2*)(out + (size_t)row * D_MODEL))[threadIdx.x] = o;
}

// ---------------------------------------------------------------------------
// bf16 GEMM: C[M,N] = A[M,K] @ B[K,N] + bias, templated epilogue.
//   MODE 0: bf16 out            (QKV projections)
//   MODE 1: bf16 out + ReLU     (FFN1)
//   MODE 2: f32 out + residual  (attn out-proj, FFN2)
// Block = 256 threads (8 waves), tile 128(M) x 128(N), K-step 32.
// Wave tile 32x64 = 2x4 WMMA tiles. Double-buffered LDS, 1 barrier/step.
// LDS row stride 34 bf16 (17 dwords) -> bank-conflict-free.
// ---------------------------------------------------------------------------
template <int MODE>
__global__ __launch_bounds__(256)
void gemm_bf16_kernel(const unsigned short* __restrict__ A,
                      const unsigned short* __restrict__ Bw,
                      const float* __restrict__ bias,
                      const float* __restrict__ residual,
                      float* __restrict__ outF,
                      unsigned short* __restrict__ outB,
                      int M, int N, int K) {
  __shared__ unsigned short Ash[2][128 * 34];   // [m][k] + pad
  __shared__ unsigned short Bts[2][128 * 34];   // transposed: [n][k] + pad
  const int tid  = threadIdx.x;
  const int lane = tid & 31;
  const int wave = tid >> 5;
  const int half = lane >> 4;                   // lane group 0/1
  const int l16  = lane & 15;
  const int mw   = (wave & 3) * 32;             // wave M offset in tile
  const int nw   = (wave >> 2) * 64;            // wave N offset in tile
  const int mblk = blockIdx.y * 128;
  const int nblk = blockIdx.x * 128;
  (void)M;

  const v8f vz = {0.f, 0.f, 0.f, 0.f, 0.f, 0.f, 0.f, 0.f};
  v8f acc[2][4];
#pragma unroll
  for (int mi = 0; mi < 2; ++mi)
#pragma unroll
    for (int ni = 0; ni < 4; ++ni) acc[mi][ni] = vz;

  union Frag { v16bf v; unsigned int w[8]; };

  // Stage A tile: 128x32 bf16 as 2048 dword chunks (coalesced along K).
  auto stageA = [&](int buf, int kb) {
#pragma unroll
    for (int p = 0; p < 8; ++p) {
      int c   = p * 256 + tid;
      int row = c >> 4;
      int kp  = (c & 15) * 2;
      *(unsigned int*)(&Ash[buf][row * 34 + kp]) =
          *(const unsigned int*)(A + (size_t)(mblk + row) * K + kb + kp);
    }
  };
  // Stage B transposed: Bts[n][k] = Bw[kb+k][nblk+n], dword global loads.
  auto stageB = [&](int buf, int kb) {
#pragma unroll
    for (int p = 0; p < 8; ++p) {
      int c = p * 256 + tid;
      int n = (c & 63) * 2;
      int k = c >> 6;
      unsigned int d =
          *(const unsigned int*)(Bw + (size_t)(kb + k) * N + nblk + n);
      Bts[buf][n * 34 + k]       = (unsigned short)(d & 0xFFFFu);
      Bts[buf][(n + 1) * 34 + k] = (unsigned short)(d >> 16);
    }
  };

  stageA(0, 0);
  stageB(0, 0);
  __syncthreads();

  for (int kb = 0; kb < K; kb += 32) {
    const int cur = (kb >> 5) & 1;
    if (kb + 32 < K) {            // stage next tile into the other buffer
      stageA(cur ^ 1, kb + 32);
      stageB(cur ^ 1, kb + 32);
    }
    if (kb + 64 < K) {            // prefetch the K+2 tile (global_prefetch_b8)
      if (tid < 128)
        __builtin_prefetch(A + (size_t)(mblk + tid) * K + kb + 64, 0, 1);
      if (tid < 32)
        __builtin_prefetch(Bw + (size_t)(kb + 64 + tid) * N + nblk, 0, 1);
    }

    // A fragments: 16-bit A 16x32 layout; lanes<16: K={0..7,16..23}, +8 hi.
    Frag af[2];
#pragma unroll
    for (int mi = 0; mi < 2; ++mi) {
      const unsigned short* ap =
          &Ash[cur][(mw + mi * 16 + l16) * 34 + half * 8];
      af[mi].w[0] = *(const unsigned int*)(ap + 0);
      af[mi].w[1] = *(const unsigned int*)(ap + 2);
      af[mi].w[2] = *(const unsigned int*)(ap + 4);
      af[mi].w[3] = *(const unsigned int*)(ap + 6);
      af[mi].w[4] = *(const unsigned int*)(ap + 16);
      af[mi].w[5] = *(const unsigned int*)(ap + 18);
      af[mi].w[6] = *(const unsigned int*)(ap + 20);
      af[mi].w[7] = *(const unsigned int*)(ap + 22);
    }
    // B fragments reused across both M sub-tiles: 8 WMMAs per K-step.
#pragma unroll
    for (int ni = 0; ni < 4; ++ni) {
      Frag bf;
      const unsigned short* bp =
          &Bts[cur][(nw + ni * 16 + l16) * 34 + half * 8];
#pragma unroll
      for (int j = 0; j < 8; ++j)
        bf.w[j] = *(const unsigned int*)(bp + (j & 3) * 2 + (j >> 2) * 16);
      acc[0][ni] = wmma_bf16(af[0].v, bf.v, acc[0][ni]);
      acc[1][ni] = wmma_bf16(af[1].v, bf.v, acc[1][ni]);
    }
    __syncthreads();
  }

  // Epilogue (specialized at compile time; C row = r + (lane>=16 ? 8 : 0)).
#pragma unroll
  for (int ni = 0; ni < 4; ++ni) {
    const int col = nblk + nw + ni * 16 + l16;
    const float bv = bias[col];
#pragma unroll
    for (int mi = 0; mi < 2; ++mi) {
#pragma unroll
      for (int r = 0; r < 8; ++r) {
        const int row = mblk + mw + mi * 16 + half * 8 + r;
        const size_t idx = (size_t)row * N + col;
        float v = acc[mi][ni][r] + bv;
        if (MODE == 1) v = fmaxf(v, 0.f);
        if (MODE == 2)
          outF[idx] = v + residual[idx];
        else
          outB[idx] = f2bf(v);
      }
    }
  }
}

// ---------------------------------------------------------------------------
// Flash-style attention. Block = 8 waves = 128 query rows of one (b,h).
// Wave owns 16 q-rows; key blocks of 32, double-buffered in LDS (1 barrier
// per iteration). P transpose goes through per-wave LDS; LDS ops from one
// wave are serviced in order (ISA 7.1), so no barrier is needed for it.
// scores: 4 WMMAs/iter, row-sum: 1 WMMA (P @ ones), ctx: 4 WMMAs/iter.
// Only the row-max reduction uses ds_bpermute shuffles.
// ---------------------------------------------------------------------------
__global__ __launch_bounds__(256)
void attn_kernel(const unsigned short* __restrict__ Qb,
                 const unsigned short* __restrict__ Kb,
                 const unsigned short* __restrict__ Vb,
                 const int* __restrict__ mask,
                 unsigned short* __restrict__ ctx) {
  __shared__ unsigned short Ksh[2][32 * 66];    // K tile [key][d], stride 66
  __shared__ unsigned short Vts[2][64 * 34];    // V tile transposed [d][key]
  __shared__ unsigned short Psh[8][16 * 34];    // per-wave P transpose buffer

  const int tid  = threadIdx.x;
  const int lane = tid & 31;
  const int wave = tid >> 5;
  const int half = lane >> 4;
  const int l16  = lane & 15;
  const int b    = blockIdx.y >> 4;
  const int h    = blockIdx.y & 15;
  const int qbase = blockIdx.x * 128 + wave * 16;

  union Frag { v16bf v; unsigned int w[8]; };

  // All-ones bf16 B fragment: row-sums on the matrix pipe (P @ ones).
  Frag ones;
#pragma unroll
  for (int j = 0; j < 8; ++j) ones.w[j] = 0x3F803F80u;   // {1.0bf, 1.0bf}

  // Q fragments (A layout, contraction over d; two 32-wide d chunks).
  Frag qf[2];
  {
    const unsigned short* qrow =
        Qb + ((size_t)(b * SEQ + qbase + l16)) * D_MODEL + h * DK;
#pragma unroll
    for (int dh = 0; dh < 2; ++dh) {
      const unsigned short* qp = qrow + dh * 32 + half * 8;
      qf[dh].w[0] = *(const unsigned int*)(qp + 0);
      qf[dh].w[1] = *(const unsigned int*)(qp + 2);
      qf[dh].w[2] = *(const unsigned int*)(qp + 4);
      qf[dh].w[3] = *(const unsigned int*)(qp + 6);
      qf[dh].w[4] = *(const unsigned int*)(qp + 16);
      qf[dh].w[5] = *(const unsigned int*)(qp + 18);
      qf[dh].w[6] = *(const unsigned int*)(qp + 20);
      qf[dh].w[7] = *(const unsigned int*)(qp + 22);
    }
  }

  // Stage K[32x64] (dword chunks) and V transposed (dword load, 2 b16 stores).
  auto stageKV = [&](int buf, int kb) {
#pragma unroll
    for (int p = 0; p < 4; ++p) {
      int c   = p * 256 + tid;
      int key = c >> 5;
      int dp  = (c & 31) * 2;
      *(unsigned int*)(&Ksh[buf][key * 66 + dp]) = *(const unsigned int*)(
          Kb + ((size_t)(b * SEQ + kb + key)) * D_MODEL + h * DK + dp);
    }
#pragma unroll
    for (int p = 0; p < 4; ++p) {
      int c   = p * 256 + tid;
      int key = c >> 5;
      int dp  = (c & 31) * 2;
      unsigned int d = *(const unsigned int*)(
          Vb + ((size_t)(b * SEQ + kb + key)) * D_MODEL + h * DK + dp);
      Vts[buf][dp * 34 + key]       = (unsigned short)(d & 0xFFFFu);
      Vts[buf][(dp + 1) * 34 + key] = (unsigned short)(d >> 16);
    }
  };

  const v8f vz = {0.f, 0.f, 0.f, 0.f, 0.f, 0.f, 0.f, 0.f};
  float m[8], l[8];
  v8f acc[4];
#pragma unroll
  for (int r = 0; r < 8; ++r) { m[r] = -3.0e38f; l[r] = 0.f; }
#pragma unroll
  for (int dt = 0; dt < 4; ++dt) acc[dt] = vz;

  stageKV(0, 0);
  __syncthreads();

  for (int kb = 0; kb < SEQ; kb += 32) {
    const int cur = (kb >> 5) & 1;
    if (kb + 32 < SEQ) stageKV(cur ^ 1, kb + 32);

    // Scores: two 16-key sub-tiles, contraction over d (2 WMMAs each).
    v8f s[2];
#pragma unroll
    for (int t = 0; t < 2; ++t) {
      s[t] = vz;
#pragma unroll
      for (int dh = 0; dh < 2; ++dh) {
        Frag kf;
        const unsigned short* kp =
            &Ksh[cur][(t * 16 + l16) * 66 + dh * 32 + half * 8];
#pragma unroll
        for (int j = 0; j < 8; ++j)
          kf.w[j] = *(const unsigned int*)(kp + (j & 3) * 2 + (j >> 2) * 16);
        s[t] = wmma_bf16(qf[dh].v, kf.v, s[t]);
      }
    }

    // Scale 1/sqrt(Dk)=0.125, apply mask (mask==0 -> -1e9).
    float ps[2][8];
#pragma unroll
    for (int t = 0; t < 2; ++t) {
      const int mv = mask[b * SEQ + kb + t * 16 + l16];
#pragma unroll
      for (int r = 0; r < 8; ++r)
        ps[t][r] = mv ? s[t][r] * 0.125f : -1.0e9f;
    }

    // Row max across the 16 key columns (lanes within each half).
    float mt[8];
#pragma unroll
    for (int r = 0; r < 8; ++r) mt[r] = fmaxf(ps[0][r], ps[1][r]);
#pragma unroll
    for (int off = 1; off < 16; off <<= 1)
#pragma unroll
      for (int r = 0; r < 8; ++r)
        mt[r] = fmaxf(mt[r], __shfl_xor(mt[r], off, 32));

    float corr[8];
#pragma unroll
    for (int r = 0; r < 8; ++r) {
      float mn = fmaxf(m[r], mt[r]);
      corr[r] = __expf(m[r] - mn);
      m[r] = mn;
    }

    // Probabilities -> bf16, transposed through LDS into A layout.
    unsigned short* pw = Psh[wave];
#pragma unroll
    for (int t = 0; t < 2; ++t)
#pragma unroll
      for (int r = 0; r < 8; ++r) {
        float p = __expf(ps[t][r] - m[r]);
        pw[(r + half * 8) * 34 + t * 16 + l16] = f2bf(p);
      }
    // Rescale running accumulators while the P stores land.
#pragma unroll
    for (int dt = 0; dt < 4; ++dt)
#pragma unroll
      for (int r = 0; r < 8; ++r) acc[dt][r] *= corr[r];

    // P as A fragment (wave-private LDS; same-wave DS ops are in order).
    Frag pf;
    const unsigned short* pp = pw + l16 * 34 + half * 8;
    pf.w[0] = *(const unsigned int*)(pp + 0);
    pf.w[1] = *(const unsigned int*)(pp + 2);
    pf.w[2] = *(const unsigned int*)(pp + 4);
    pf.w[3] = *(const unsigned int*)(pp + 6);
    pf.w[4] = *(const unsigned int*)(pp + 16);
    pf.w[5] = *(const unsigned int*)(pp + 18);
    pf.w[6] = *(const unsigned int*)(pp + 20);
    pf.w[7] = *(const unsigned int*)(pp + 22);

    // Row sums on the matrix pipe: rs = P @ ones(32x16); every lane of a
    // half holds its rows' sums in the C layout (row = r + half*8).
    v8f rst = wmma_bf16(pf.v, ones.v, vz);
#pragma unroll
    for (int r = 0; r < 8; ++r) l[r] = l[r] * corr[r] + rst[r];

    // ctx += P @ V  (4 d-tiles of 16).
#pragma unroll
    for (int dt = 0; dt < 4; ++dt) {
      Frag vf;
      const unsigned short* vp =
          &Vts[cur][(dt * 16 + l16) * 34 + half * 8];
#pragma unroll
      for (int j = 0; j < 8; ++j)
        vf.w[j] = *(const unsigned int*)(vp + (j & 3) * 2 + (j >> 2) * 16);
      acc[dt] = wmma_bf16(pf.v, vf.v, acc[dt]);
    }
    __syncthreads();
  }

  // Normalize and write ctx (bf16, [B,S,H*Dk] layout for output projection).
#pragma unroll
  for (int dt = 0; dt < 4; ++dt)
#pragma unroll
    for (int r = 0; r < 8; ++r) {
      const int row = qbase + half * 8 + r;
      float v = acc[dt][r] / l[r];
      ctx[((size_t)(b * SEQ + row)) * D_MODEL + h * DK + dt * 16 + l16] =
          f2bf(v);
    }
}

// ---------------------------------------------------------------------------
// Launch
// ---------------------------------------------------------------------------
extern "C" void kernel_launch(void* const* d_in, const int* in_sizes, int n_in,
                              void* d_out, int out_size, void* d_ws,
                              size_t ws_size, hipStream_t stream) {
  (void)in_sizes; (void)n_in; (void)out_size; (void)ws_size;
  const float* x    = (const float*)d_in[0];
  const int*   mask = (const int*)d_in[1];
  const float* wq   = (const float*)d_in[2];
  const float* bq   = (const float*)d_in[3];
  const float* wk   = (const float*)d_in[4];
  const float* bk   = (const float*)d_in[5];
  const float* wv   = (const float*)d_in[6];
  const float* bv   = (const float*)d_in[7];
  const float* wo   = (const float*)d_in[8];
  const float* bo   = (const float*)d_in[9];
  const float* w1   = (const float*)d_in[10];
  const float* b1   = (const float*)d_in[11];
  const float* w2   = (const float*)d_in[12];
  const float* b2   = (const float*)d_in[13];
  const float* ln1a = (const float*)d_in[14];
  const float* ln1b = (const float*)d_in[15];
  const float* ln2a = (const float*)d_in[16];
  const float* ln2b = (const float*)d_in[17];
  float* out = (float*)d_out;

  char* ws = (char*)d_ws;
  size_t off = 0;
  auto alloc = [&](size_t bytes) -> void* {
    void* p = ws + off;
    off += (bytes + 255) & ~(size_t)255;
    return p;
  };
  unsigned short* wq_b  = (unsigned short*)alloc((size_t)D_MODEL * D_MODEL * 2);
  unsigned short* wk_b  = (unsigned short*)alloc((size_t)D_MODEL * D_MODEL * 2);
  unsigned short* wv_b  = (unsigned short*)alloc((size_t)D_MODEL * D_MODEL * 2);
  unsigned short* wo_b  = (unsigned short*)alloc((size_t)D_MODEL * D_MODEL * 2);
  unsigned short* w1_b  = (unsigned short*)alloc((size_t)D_MODEL * D_FF * 2);
  unsigned short* w2_b  = (unsigned short*)alloc((size_t)D_FF * D_MODEL * 2);
  unsigned short* xn_b  = (unsigned short*)alloc((size_t)ROWS * D_MODEL * 2);
  unsigned short* q_b   = (unsigned short*)alloc((size_t)ROWS * D_MODEL * 2);
  unsigned short* k_b   = (unsigned short*)alloc((size_t)ROWS * D_MODEL * 2);
  unsigned short* v_b   = (unsigned short*)alloc((size_t)ROWS * D_MODEL * 2);
  unsigned short* ctx_b = (unsigned short*)alloc((size_t)ROWS * D_MODEL * 2);
  float*          x1    = (float*)alloc((size_t)ROWS * D_MODEL * 4);
  unsigned short* xn2_b = (unsigned short*)alloc((size_t)ROWS * D_MODEL * 2);
  unsigned short* h_b   = (unsigned short*)alloc((size_t)ROWS * D_FF * 2);

  // Weights -> bf16.
  cast_f32_bf16_kernel<<<256, 256, 0, stream>>>(
      (const float4*)wq, (uint2*)wq_b, D_MODEL * D_MODEL / 4);
  cast_f32_bf16_kernel<<<256, 256, 0, stream>>>(
      (const float4*)wk, (uint2*)wk_b, D_MODEL * D_MODEL / 4);
  cast_f32_bf16_kernel<<<256, 256, 0, stream>>>(
      (const float4*)wv, (uint2*)wv_b, D_MODEL * D_MODEL / 4);
  cast_f32_bf16_kernel<<<256, 256, 0, stream>>>(
      (const float4*)wo, (uint2*)wo_b, D_MODEL * D_MODEL / 4);
  cast_f32_bf16_kernel<<<1024, 256, 0, stream>>>(
      (const float4*)w1, (uint2*)w1_b, D_MODEL * D_FF / 4);
  cast_f32_bf16_kernel<<<1024, 256, 0, stream>>>(
      (const float4*)w2, (uint2*)w2_b, D_FF * D_MODEL / 4);

  // LN1.
  ln_bf16_kernel<<<ROWS, 256, 0, stream>>>(x, ln1a, ln1b, xn_b);

  // Q/K/V projections (bf16 out).
  dim3 gP(D_MODEL / 128, ROWS / 128);
  gemm_bf16_kernel<0><<<gP, 256, 0, stream>>>(xn_b, wq_b, bq, nullptr, nullptr,
                                              q_b, ROWS, D_MODEL, D_MODEL);
  gemm_bf16_kernel<0><<<gP, 256, 0, stream>>>(xn_b, wk_b, bk, nullptr, nullptr,
                                              k_b, ROWS, D_MODEL, D_MODEL);
  gemm_bf16_kernel<0><<<gP, 256, 0, stream>>>(xn_b, wv_b, bv, nullptr, nullptr,
                                              v_b, ROWS, D_MODEL, D_MODEL);

  // Attention.
  dim3 gA(SEQ / 128, BATCH * NH);
  attn_kernel<<<gA, 256, 0, stream>>>(q_b, k_b, v_b, mask, ctx_b);

  // Output projection + residual -> x1 (fp32).
  gemm_bf16_kernel<2><<<gP, 256, 0, stream>>>(ctx_b, wo_b, bo, x, x1, nullptr,
                                              ROWS, D_MODEL, D_MODEL);

  // LN2.
  ln_bf16_kernel<<<ROWS, 256, 0, stream>>>(x1, ln2a, ln2b, xn2_b);

  // FFN1: relu(xn2 @ w1 + b1) -> h (bf16).
  dim3 gF1(D_FF / 128, ROWS / 128);
  gemm_bf16_kernel<1><<<gF1, 256, 0, stream>>>(xn2_b, w1_b, b1, nullptr,
                                               nullptr, h_b, ROWS, D_FF,
                                               D_MODEL);

  // FFN2: h @ w2 + b2 + x1 -> out (fp32).
  gemm_bf16_kernel<2><<<gP, 256, 0, stream>>>(h_b, w2_b, b2, x1, out, nullptr,
                                              ROWS, D_MODEL, D_FF);
}
